// Transformer_49624052138130
// MI455X (gfx1250) — compile-verified, matrix operand order checked by reading, not run on hardware
//
#include <hip/hip_runtime.h>
#include <hip/hip_bf16.h>

// ---------------------------------------------------------------------------
// Transformer forward (enc-dec) for gfx1250 / MI455X.
// All GEMMs via v_wmma_f32_16x16x32_f16 (f16 in, f32 acc).
// GEMM: 256 thr = 8 wave32, block tile 128x64, wave tile 32x32 (2x2 WMMA),
// double-buffered LDS, one barrier per K-step.
// All B operands are N-major [N][K] (weights pre-transposed at cast time;
// V's f16 shadow written transposed by the epilogue), so A and B staging are
// both contiguous b128 global->LDS copies.
// ---------------------------------------------------------------------------

typedef _Float16 v16h __attribute__((ext_vector_type(16)));
typedef _Float16 v8h  __attribute__((ext_vector_type(8)));
typedef float    v8f  __attribute__((ext_vector_type(8)));

#define D_MODEL 512
#define N_HEAD  8
#define D_HEAD  64
#define FF_DIM  2048
#define SEQ     1024
#define BATCHN  2
#define NLAYER  6
#define VOCAB   32000
#define ROWS    (BATCHN*SEQ)   // 2048 token rows

// ---------------- cast f32 -> f16 with transpose: dst[n][k] = src[k][n] ----
// src: [Z][Kd][Nd] (blockIdx.z selects slab), dst: [Z][Nd][Kd]
__global__ void transpose_cast(const float* __restrict__ src,
                               _Float16* __restrict__ dst, int Kd, int Nd) {
  __shared__ float tile[32][33];
  long zoff = (long)blockIdx.z * Kd * Nd;
  int k0 = blockIdx.x * 32, n0 = blockIdx.y * 32;
  int tx = threadIdx.x & 31, ty = threadIdx.x >> 5;   // 32 x 8
  #pragma unroll
  for (int j = 0; j < 32; j += 8)
    tile[ty + j][tx] = src[zoff + (long)(k0 + ty + j) * Nd + (n0 + tx)];
  __syncthreads();
  #pragma unroll
  for (int j = 0; j < 32; j += 8)
    dst[zoff + (long)(n0 + ty + j) * Kd + (k0 + tx)] = (_Float16)tile[tx][ty + j];
}

// ------------------- embedding * sqrt(D) + positional enc ------------------
__global__ void embed_pe(const int* __restrict__ ids,
                         const float* __restrict__ table,
                         float* __restrict__ out, _Float16* __restrict__ out16,
                         int seq) {
  int row = blockIdx.x;              // 0..ROWS-1
  int pos = row % seq;
  int id  = ids[row];
  const float scale = 22.62741699796952f;  // sqrt(512)
  #pragma unroll
  for (int j = 0; j < 4; ++j) {
    int c  = threadIdx.x + 128 * j;  // blockDim = 128, D_MODEL = 512
    int ce = c & ~1;
    float ang = (float)pos * powf(10000.0f, -(float)ce / (float)D_MODEL);
    float pe  = (c & 1) ? cosf(ang) : sinf(ang);
    float v   = table[(long)id * D_MODEL + c] * scale + pe;
    long  o   = (long)row * D_MODEL + c;
    out[o]   = v;
    out16[o] = (_Float16)v;
  }
}

// --------------------- fused residual add + LayerNorm ----------------------
__global__ void add_layernorm(const float* __restrict__ X,
                              const float* __restrict__ Y,
                              const float* __restrict__ g,
                              const float* __restrict__ b,
                              float* __restrict__ O,
                              _Float16* __restrict__ O16) {
  __shared__ float red[128];
  int  row  = blockIdx.x;
  long base = (long)row * D_MODEL;
  float v[4];
  float s = 0.f;
  #pragma unroll
  for (int j = 0; j < 4; ++j) {
    int c = threadIdx.x + 128 * j;
    v[j] = X[base + c] + Y[base + c];
    s += v[j];
  }
  red[threadIdx.x] = s; __syncthreads();
  for (int off = 64; off > 0; off >>= 1) {
    if (threadIdx.x < off) red[threadIdx.x] += red[threadIdx.x + off];
    __syncthreads();
  }
  float mu = red[0] * (1.0f / D_MODEL);
  __syncthreads();
  float s2 = 0.f;
  #pragma unroll
  for (int j = 0; j < 4; ++j) { float d = v[j] - mu; s2 += d * d; }
  red[threadIdx.x] = s2; __syncthreads();
  for (int off = 64; off > 0; off >>= 1) {
    if (threadIdx.x < off) red[threadIdx.x] += red[threadIdx.x + off];
    __syncthreads();
  }
  float rstd = rsqrtf(red[0] * (1.0f / D_MODEL) + 1e-6f);
  #pragma unroll
  for (int j = 0; j < 4; ++j) {
    int c = threadIdx.x + 128 * j;
    float o = (v[j] - mu) * rstd * g[c] + b[c];
    O[base + c]   = o;
    O16[base + c] = (_Float16)o;
  }
}

// ------------------ masked softmax over rows of length SEQ -----------------
__global__ void softmax_mask(const float* __restrict__ S,
                             const float* __restrict__ mask,
                             _Float16* __restrict__ P) {
  __shared__ float red[256];
  long base = (long)blockIdx.x * SEQ;
  float v[4];
  float mx = -3.0e38f;
  #pragma unroll
  for (int j = 0; j < 4; ++j) {
    int c = threadIdx.x + 256 * j;
    float x = S[base + c] + mask[base + c] * (-1e9f);
    v[j] = x; mx = fmaxf(mx, x);
  }
  red[threadIdx.x] = mx; __syncthreads();
  for (int off = 128; off > 0; off >>= 1) {
    if (threadIdx.x < off) red[threadIdx.x] = fmaxf(red[threadIdx.x], red[threadIdx.x + off]);
    __syncthreads();
  }
  mx = red[0]; __syncthreads();
  float s = 0.f;
  #pragma unroll
  for (int j = 0; j < 4; ++j) { v[j] = __expf(v[j] - mx); s += v[j]; }
  red[threadIdx.x] = s; __syncthreads();
  for (int off = 128; off > 0; off >>= 1) {
    if (threadIdx.x < off) red[threadIdx.x] += red[threadIdx.x + off];
    __syncthreads();
  }
  float inv = 1.0f / red[0];
  #pragma unroll
  for (int j = 0; j < 4; ++j) {
    int c = threadIdx.x + 256 * j;
    P[base + c] = (_Float16)(v[j] * inv);
  }
}

// ------------------------------ WMMA GEMM ----------------------------------
// C = alpha * A[M,K] * B + bias (+relu).  A f16 row-major (lda).
// B f16 N-major: element(k,n) = B[n*ldb + k].
// MODE: 0 -> f32 C[m*ldc+n];  1 -> f16 C[m*ldc+n];  2 -> f16 C[n*ldc+m]
// (transposed shadow; lane's 8 M-rows are contiguous -> one b128 store).
// Batched over blockIdx.z: z -> bb=z/bh, hh=z%bh with split strides.
// Requires: M%128==0, N%64==0, K%32==0, lda/ldb multiples of 8.
template <int MODE>
__global__ void __launch_bounds__(256)
gemm_wmma_f16(const _Float16* __restrict__ A, const _Float16* __restrict__ B,
              const float* __restrict__ bias, void* __restrict__ Cout,
              int M, int N, int K, int lda, int ldb, int ldc,
              long sAb, long sAh, long sBb, long sBh, long sCb, long sCh,
              int bh, float alpha, int relu) {
  __shared__ _Float16 As[2][128][40];  // 128 rows x 32 K, pad 8 -> 80B rows
  __shared__ _Float16 Bs[2][64][40];   // 64 cols x 32 K (N-major)

  int z  = blockIdx.z;
  int bb = z / bh, hh = z % bh;
  A += bb * sAb + hh * sAh;
  B += bb * sBb + hh * sBh;
  long cbase = bb * sCb + hh * sCh;

  int n0 = blockIdx.x * 64, m0 = blockIdx.y * 128;
  int t = threadIdx.x;
  int lane = t & 31, wave = t >> 5;
  int waveM = wave >> 1, waveN = wave & 1;       // 4x2 waves of 32x32
  int rl = lane & 15, kh = (lane >> 4) * 8;      // CDNA5 16-bit frag layout

  // cooperative-load indices: all contiguous 16B chunks
  int ar = t >> 1, ac = (t & 1) * 16;            // A: 128 rows x 32 halves
  int bn = t >> 2, bc = (t & 3) * 8;             // B: 64 rows x 32 halves

  v8h ax, ay, bx;
  auto fetch = [&](int k0) {
    const _Float16* pA = &A[(long)(m0 + ar) * lda + k0 + ac];
    ax = *(const v8h*)pA;
    ay = *(const v8h*)(pA + 8);
    bx = *(const v8h*)&B[(long)(n0 + bn) * ldb + k0 + bc];
  };
  auto stage = [&](int buf) {
    *(v8h*)&As[buf][ar][ac]     = ax;
    *(v8h*)&As[buf][ar][ac + 8] = ay;
    *(v8h*)&Bs[buf][bn][bc]     = bx;
  };

  v8f zero = {};
  v8f acc[2][2] = {{zero, zero}, {zero, zero}};

  fetch(0);
  stage(0);
  int buf = 0;
  for (int k0 = 0; k0 < K; k0 += 32) {
    bool more = (k0 + 32) < K;
    if (more) fetch(k0 + 32);        // overlap global latency with WMMA
    __syncthreads();                 // stage(buf) visible; buf^1 reads done
    union { v16h v; v8h h[2]; } aF[2], bF[2];
    #pragma unroll
    for (int i = 0; i < 2; ++i) {
      int rr = waveM * 32 + i * 16 + rl;
      aF[i].h[0] = *(const v8h*)&As[buf][rr][kh];
      aF[i].h[1] = *(const v8h*)&As[buf][rr][kh + 16];
      int cc = waveN * 32 + i * 16 + rl;
      bF[i].h[0] = *(const v8h*)&Bs[buf][cc][kh];
      bF[i].h[1] = *(const v8h*)&Bs[buf][cc][kh + 16];
    }
    #pragma unroll
    for (int i = 0; i < 2; ++i)
      #pragma unroll
      for (int j = 0; j < 2; ++j)
        acc[i][j] = __builtin_amdgcn_wmma_f32_16x16x32_f16(
            false, aF[i].v, false, bF[j].v, (short)0, acc[i][j], false, false);
    if (more) { stage(buf ^ 1); buf ^= 1; }  // other buffer: no conflict
  }

  // Epilogue. C frag: VGPR q holds M = q + 8*(lane>=16), N = lane&15.
  #pragma unroll
  for (int i = 0; i < 2; ++i) {
    #pragma unroll
    for (int j = 0; j < 2; ++j) {
      int col  = n0 + waveN * 32 + j * 16 + rl;
      int row0 = m0 + waveM * 32 + i * 16 + (lane >> 4) * 8;
      float bv = bias ? bias[col] : 0.0f;
      if (MODE == 2) {
        v8h pk;
        #pragma unroll
        for (int q = 0; q < 8; ++q) {
          float x = acc[i][j][q] * alpha + bv;
          if (relu) x = fmaxf(x, 0.0f);
          pk[q] = (_Float16)x;
        }
        *(v8h*)((_Float16*)Cout + cbase + (long)col * ldc + row0) = pk;
      } else {
        #pragma unroll
        for (int q = 0; q < 8; ++q) {
          float x = acc[i][j][q] * alpha + bv;
          if (relu) x = fmaxf(x, 0.0f);
          long off = cbase + (long)(row0 + q) * ldc + col;
          if (MODE == 0) ((float*)Cout)[off] = x;
          else           ((_Float16*)Cout)[off] = (_Float16)x;
        }
      }
    }
  }
}

// ---------------------------------------------------------------------------
extern "C" void kernel_launch(void* const* d_in, const int* in_sizes, int n_in,
                              void* d_out, int out_size, void* d_ws, size_t ws_size,
                              hipStream_t stream) {
  (void)in_sizes; (void)n_in; (void)out_size; (void)ws_size;

  // -------- inputs in setup_inputs() dict (insertion) order --------
  const int*   ids_in  = (const int*)d_in[0];
  const int*   ids_tg  = (const int*)d_in[1];
  const float* m_inp   = (const float*)d_in[2];   // (B,H,S,S) zeros
  const float* m_la    = (const float*)d_in[3];   // look-ahead
  const float* m_tp    = (const float*)d_in[4];   // target padding
  const float* emb_in  = (const float*)d_in[5];
  const float* emb_out = (const float*)d_in[6];
  const float* E_wq  = (const float*)d_in[7],  *E_bq  = (const float*)d_in[8];
  const float* E_wk  = (const float*)d_in[9],  *E_bk  = (const float*)d_in[10];
  const float* E_wv  = (const float*)d_in[11], *E_bv  = (const float*)d_in[12];
  const float* E_wo  = (const float*)d_in[13], *E_bo  = (const float*)d_in[14];
  const float* E_ln1g= (const float*)d_in[15], *E_ln1b= (const float*)d_in[16];
  const float* E_w1  = (const float*)d_in[17], *E_b1  = (const float*)d_in[18];
  const float* E_w2  = (const float*)d_in[19], *E_b2  = (const float*)d_in[20];
  const float* E_ln2g= (const float*)d_in[21], *E_ln2b= (const float*)d_in[22];
  const float* A1_wq = (const float*)d_in[23], *A1_bq = (const float*)d_in[24];
  const float* A1_wk = (const float*)d_in[25], *A1_bk = (const float*)d_in[26];
  const float* A1_wv = (const float*)d_in[27], *A1_bv = (const float*)d_in[28];
  const float* A1_wo = (const float*)d_in[29], *A1_bo = (const float*)d_in[30];
  const float* A2_wq = (const float*)d_in[31], *A2_bq = (const float*)d_in[32];
  const float* A2_wk = (const float*)d_in[33], *A2_bk = (const float*)d_in[34];
  const float* A2_wv = (const float*)d_in[35], *A2_bv = (const float*)d_in[36];
  const float* A2_wo = (const float*)d_in[37], *A2_bo = (const float*)d_in[38];
  const float* D_ln1g= (const float*)d_in[39], *D_ln1b= (const float*)d_in[40];
  const float* D_ln2g= (const float*)d_in[41], *D_ln2b= (const float*)d_in[42];
  const float* D_ln3g= (const float*)d_in[43], *D_ln3b= (const float*)d_in[44];
  const float* D_w1  = (const float*)d_in[45], *D_b1  = (const float*)d_in[46];
  const float* D_w2  = (const float*)d_in[47], *D_b2  = (const float*)d_in[48];
  const float* FW    = (const float*)d_in[49], *FB    = (const float*)d_in[50];

  // -------- workspace carve --------
  char* p = (char*)d_ws;
  auto carve = [&](size_t bytes) -> char* {
    char* r = p; p += (bytes + 255) & ~(size_t)255; return r;
  };
  const size_t WATT = (size_t)NLAYER * D_MODEL * D_MODEL;
  const size_t WFF  = (size_t)NLAYER * D_MODEL * FF_DIM;

  // All weights stored transposed ([N][K]) in f16.
  _Float16* e_wqT = (_Float16*)carve(WATT * 2);
  _Float16* e_wkT = (_Float16*)carve(WATT * 2);
  _Float16* e_wvT = (_Float16*)carve(WATT * 2);
  _Float16* e_woT = (_Float16*)carve(WATT * 2);
  _Float16* e_w1T = (_Float16*)carve(WFF * 2);
  _Float16* e_w2T = (_Float16*)carve(WFF * 2);
  _Float16* d1_wqT = (_Float16*)carve(WATT * 2);
  _Float16* d1_wkT = (_Float16*)carve(WATT * 2);
  _Float16* d1_wvT = (_Float16*)carve(WATT * 2);
  _Float16* d1_woT = (_Float16*)carve(WATT * 2);
  _Float16* d2_wqT = (_Float16*)carve(WATT * 2);
  _Float16* d2_wkT = (_Float16*)carve(WATT * 2);
  _Float16* d2_wvT = (_Float16*)carve(WATT * 2);
  _Float16* d2_woT = (_Float16*)carve(WATT * 2);
  _Float16* d_w1T  = (_Float16*)carve(WFF * 2);
  _Float16* d_w2T  = (_Float16*)carve(WFF * 2);
  _Float16* fwT    = (_Float16*)carve((size_t)D_MODEL * VOCAB * 2);

  float*    encx    = (float*)carve((size_t)ROWS * D_MODEL * 4);
  _Float16* encx16  = (_Float16*)carve((size_t)ROWS * D_MODEL * 2);
  float*    decx    = (float*)carve((size_t)ROWS * D_MODEL * 4);
  _Float16* decx16  = (_Float16*)carve((size_t)ROWS * D_MODEL * 2);
  float*    tmpA    = (float*)carve((size_t)ROWS * D_MODEL * 4);
  _Float16* q16     = (_Float16*)carve((size_t)ROWS * D_MODEL * 2);
  _Float16* k16     = (_Float16*)carve((size_t)ROWS * D_MODEL * 2);
  _Float16* v16T    = (_Float16*)carve((size_t)ROWS * D_MODEL * 2); // [512][2048]
  _Float16* ctx16   = (_Float16*)carve((size_t)ROWS * D_MODEL * 2);
  float*    scores  = (float*)carve((size_t)BATCHN * N_HEAD * SEQ * SEQ * 4);
  _Float16* probs16 = (_Float16*)carve((size_t)BATCHN * N_HEAD * SEQ * SEQ * 2);
  _Float16* ffh16   = (_Float16*)carve((size_t)ROWS * FF_DIM * 2);

  // -------- helpers --------
  auto tcast = [&](const float* s, _Float16* d, int Kd, int Nd, int Z) {
    dim3 g(Kd / 32, Nd / 32, Z);
    transpose_cast<<<g, 256, 0, stream>>>(s, d, Kd, Nd);
  };
  auto gemm = [&](const _Float16* Aa, const _Float16* Bb, const float* bias,
                  void* Cc, int mode, int M, int N, int K,
                  int lda, int ldb, int ldc,
                  long sAb, long sAh, long sBb, long sBh, long sCb, long sCh,
                  int Z, int bh, float alpha, int relu) {
    dim3 g(N / 64, M / 128, Z);
    if (mode == 0)
      gemm_wmma_f16<0><<<g, 256, 0, stream>>>(Aa, Bb, bias, Cc, M, N, K,
          lda, ldb, ldc, sAb, sAh, sBb, sBh, sCb, sCh, bh, alpha, relu);
    else if (mode == 1)
      gemm_wmma_f16<1><<<g, 256, 0, stream>>>(Aa, Bb, bias, Cc, M, N, K,
          lda, ldb, ldc, sAb, sAh, sBb, sBh, sCb, sCh, bh, alpha, relu);
    else
      gemm_wmma_f16<2><<<g, 256, 0, stream>>>(Aa, Bb, bias, Cc, M, N, K,
          lda, ldb, ldc, sAb, sAh, sBb, sBh, sCb, sCh, bh, alpha, relu);
  };
  auto lnorm = [&](const float* X, const float* Y, const float* g,
                   const float* b, float* O, _Float16* O16) {
    add_layernorm<<<ROWS, 128, 0, stream>>>(X, Y, g, b, O, O16);
  };
  const long SD = (long)SEQ * D_MODEL;
  const long SS = (long)SEQ * SEQ;
  auto attention = [&](const _Float16* xq16, const _Float16* xkv16,
                       const _Float16* wqT, const float* bq,
                       const _Float16* wkT, const float* bk,
                       const _Float16* wvT, const float* bv,
                       const _Float16* woT, const float* bo,
                       const float* mask, float* outA) {
    gemm(xq16,  wqT, bq, q16, 1, ROWS, D_MODEL, D_MODEL, D_MODEL, D_MODEL,
         D_MODEL, 0,0,0,0,0,0, 1,1, 1.f, 0);
    gemm(xkv16, wkT, bk, k16, 1, ROWS, D_MODEL, D_MODEL, D_MODEL, D_MODEL,
         D_MODEL, 0,0,0,0,0,0, 1,1, 1.f, 0);
    // V projection: f16 shadow written transposed -> v16T [D_MODEL][ROWS]
    gemm(xkv16, wvT, bv, v16T, 2, ROWS, D_MODEL, D_MODEL, D_MODEL, D_MODEL,
         ROWS, 0,0,0,0,0,0, 1,1, 1.f, 0);
    // scores[b,h] = Q_h * K_h^T / d_k   (M=S, N=S, K=64); B = k16 (N-major)
    gemm(q16, k16, nullptr, scores, 0, SEQ, SEQ, D_HEAD,
         D_MODEL, D_MODEL, SEQ,
         SD, D_HEAD, SD, D_HEAD, (long)N_HEAD * SS, SS,
         BATCHN * N_HEAD, N_HEAD, 1.0f / (float)D_HEAD, 0);
    softmax_mask<<<BATCHN * N_HEAD * SEQ, 256, 0, stream>>>(scores, mask, probs16);
    // ctx[b,h] = P * V_h  (M=S, N=64, K=S); B = v16T (N-major, ldb=ROWS)
    gemm(probs16, v16T, nullptr, ctx16, 1, SEQ, D_HEAD, SEQ,
         SEQ, ROWS, D_MODEL,
         (long)N_HEAD * SS, SS, (long)SEQ, (long)D_HEAD * ROWS, SD, D_HEAD,
         BATCHN * N_HEAD, N_HEAD, 1.f, 0);
    gemm(ctx16, woT, bo, outA, 0, ROWS, D_MODEL, D_MODEL, D_MODEL, D_MODEL,
         D_MODEL, 0,0,0,0,0,0, 1,1, 1.f, 0);
  };
  auto ffn = [&](const _Float16* x16, const _Float16* w1T, const float* b1,
                 const _Float16* w2T, const float* b2, float* outA) {
    gemm(x16, w1T, b1, ffh16, 1, ROWS, FF_DIM, D_MODEL, D_MODEL, D_MODEL,
         FF_DIM, 0,0,0,0,0,0, 1,1, 1.f, 1);
    gemm(ffh16, w2T, b2, outA, 0, ROWS, D_MODEL, FF_DIM, FF_DIM, FF_DIM,
         D_MODEL, 0,0,0,0,0,0, 1,1, 1.f, 0);
  };

  // -------- weight conversion: f32 -> f16, transposed to [N][K] --------
  tcast(E_wq, e_wqT, D_MODEL, D_MODEL, NLAYER);
  tcast(E_wk, e_wkT, D_MODEL, D_MODEL, NLAYER);
  tcast(E_wv, e_wvT, D_MODEL, D_MODEL, NLAYER);
  tcast(E_wo, e_woT, D_MODEL, D_MODEL, NLAYER);
  tcast(E_w1, e_w1T, D_MODEL, FF_DIM, NLAYER);
  tcast(E_w2, e_w2T, FF_DIM, D_MODEL, NLAYER);
  tcast(A1_wq, d1_wqT, D_MODEL, D_MODEL, NLAYER);
  tcast(A1_wk, d1_wkT, D_MODEL, D_MODEL, NLAYER);
  tcast(A1_wv, d1_wvT, D_MODEL, D_MODEL, NLAYER);
  tcast(A1_wo, d1_woT, D_MODEL, D_MODEL, NLAYER);
  tcast(A2_wq, d2_wqT, D_MODEL, D_MODEL, NLAYER);
  tcast(A2_wk, d2_wkT, D_MODEL, D_MODEL, NLAYER);
  tcast(A2_wv, d2_wvT, D_MODEL, D_MODEL, NLAYER);
  tcast(A2_wo, d2_woT, D_MODEL, D_MODEL, NLAYER);
  tcast(D_w1, d_w1T, D_MODEL, FF_DIM, NLAYER);
  tcast(D_w2, d_w2T, FF_DIM, D_MODEL, NLAYER);
  tcast(FW, fwT, D_MODEL, VOCAB, 1);

  const size_t PA = (size_t)D_MODEL * D_MODEL;
  const size_t PF = (size_t)D_MODEL * FF_DIM;

  // -------- encoder --------
  embed_pe<<<ROWS, 128, 0, stream>>>(ids_in, emb_in, encx, encx16, SEQ);
  for (int l = 0; l < NLAYER; ++l) {
    attention(encx16, encx16,
              e_wqT + l * PA, E_bq + l * D_MODEL,
              e_wkT + l * PA, E_bk + l * D_MODEL,
              e_wvT + l * PA, E_bv + l * D_MODEL,
              e_woT + l * PA, E_bo + l * D_MODEL,
              m_inp, tmpA);
    lnorm(encx, tmpA, E_ln1g + l * D_MODEL, E_ln1b + l * D_MODEL, encx, encx16);
    ffn(encx16, e_w1T + l * PF, E_b1 + l * FF_DIM,
        e_w2T + l * PF, E_b2 + l * D_MODEL, tmpA);
    lnorm(encx, tmpA, E_ln2g + l * D_MODEL, E_ln2b + l * D_MODEL, encx, encx16);
  }

  // -------- decoder --------
  embed_pe<<<ROWS, 128, 0, stream>>>(ids_tg, emb_out, decx, decx16, SEQ);
  for (int l = 0; l < NLAYER; ++l) {
    attention(decx16, decx16,
              d1_wqT + l * PA, A1_bq + l * D_MODEL,
              d1_wkT + l * PA, A1_bk + l * D_MODEL,
              d1_wvT + l * PA, A1_bv + l * D_MODEL,
              d1_woT + l * PA, A1_bo + l * D_MODEL,
              m_la, tmpA);
    lnorm(decx, tmpA, D_ln1g + l * D_MODEL, D_ln1b + l * D_MODEL, decx, decx16);
    attention(decx16, encx16,
              d2_wqT + l * PA, A2_bq + l * D_MODEL,
              d2_wkT + l * PA, A2_bk + l * D_MODEL,
              d2_wvT + l * PA, A2_bv + l * D_MODEL,
              d2_woT + l * PA, A2_bo + l * D_MODEL,
              m_tp, tmpA);
    lnorm(decx, tmpA, D_ln2g + l * D_MODEL, D_ln2b + l * D_MODEL, decx, decx16);
    ffn(decx16, d_w1T + l * PF, D_b1 + l * FF_DIM,
        d_w2T + l * PF, D_b2 + l * D_MODEL, tmpA);
    lnorm(decx, tmpA, D_ln3g + l * D_MODEL, D_ln3b + l * D_MODEL, decx, decx16);
  }

  // -------- final vocab projection (dominant GEMM: 2048x32000x512) --------
  gemm(decx16, fwT, FB, d_out, 0, ROWS, VOCAB, D_MODEL,
       D_MODEL, D_MODEL, VOCAB, 0,0,0,0,0,0, 1,1, 1.f, 0);
}
